// DoubleTransformer_24919400251901
// MI455X (gfx1250) — compile-verified, matrix operand order checked by reading, not run on hardware
//
#include <hip/hip_runtime.h>
#include <hip/hip_bf16.h>

// ---------------------------------------------------------------------------
// DoubleTransformer forward for MI455X (gfx1250, wave32, WMMA).
// Heavy GEMMs: v_wmma_f32_16x16x32_bf16 with fp32 accumulation.
// Global->LDS tile movement uses CDNA5 async loads (ASYNCcnt), eliminating
// VGPR staging (and the scratch spills seen in round 1).
// ---------------------------------------------------------------------------

typedef __attribute__((ext_vector_type(16))) __bf16 v16bf;
typedef __attribute__((ext_vector_type(8)))  __bf16 v8bf;
typedef __attribute__((ext_vector_type(8)))  float  v8f;
typedef __attribute__((ext_vector_type(4)))  int    v4i;

#define DMODEL 512
#define MROWS  32768        // 64 * 512 tokens
#define LDSTR  40           // padded LDS row stride (elements) for 32-wide K tiles

#if __has_builtin(__builtin_amdgcn_global_load_async_to_lds_b128)
#define USE_ASYNC_LDS 1
#else
#define USE_ASYNC_LDS 0
#endif

__device__ __forceinline__ void wait_async0() {
#if __has_builtin(__builtin_amdgcn_s_wait_asynccnt)
  __builtin_amdgcn_s_wait_asynccnt(0);
#else
  asm volatile("s_wait_asynccnt 0x0" ::: "memory");
#endif
}

#if USE_ASYNC_LDS
// one 16-byte async global->LDS copy per lane (no VGPR data staging).
// builtin signature (from round-2 diagnostic): (v4i as1*, v4i as3*, imm, imm)
__device__ __forceinline__ void async_cp16(const __bf16* g, __bf16* l) {
  __builtin_amdgcn_global_load_async_to_lds_b128(
      (__attribute__((address_space(1))) v4i*)(v4i*)g,
      (__attribute__((address_space(3))) v4i*)(v4i*)l, 0, 0);
}
#endif

// ---------------------------------------------------------------------------
// WMMA fragment load from LDS. 16-bit A/B operand layout (ISA 7.12.2):
// lanes 0-15: elems 0..7 -> K=0..7,  elems 8..15 -> K=16..23
// lanes16-31: elems 0..7 -> K=8..15, elems 8..15 -> K=24..31
// => two ds_load_b128 per fragment.
// ---------------------------------------------------------------------------
__device__ __forceinline__ v16bf load_frag(const __bf16* row_ptr, int hf) {
  v8bf lo = *(const v8bf*)(row_ptr + hf * 8);
  v8bf hi = *(const v8bf*)(row_ptr + hf * 8 + 16);
  v16bf r;
#pragma unroll
  for (int i = 0; i < 8; ++i) { r[i] = lo[i]; r[i + 8] = hi[i]; }
  return r;
}

__device__ __forceinline__ void stage_load(const __bf16* __restrict__ G, long K,
                                           long rowBase, long kbase, int t, uint4* r) {
  const int i0 = t, i1 = t + 256;   // 512 uint4 per 128x32 bf16 tile
  r[0] = *(const uint4*)(G + (rowBase + (long)(i0 >> 2)) * K + kbase + (long)(i0 & 3) * 8);
  r[1] = *(const uint4*)(G + (rowBase + (long)(i1 >> 2)) * K + kbase + (long)(i1 & 3) * 8);
}

__device__ __forceinline__ void stage_store(__bf16* s, int t, const uint4* r) {
  const int i0 = t, i1 = t + 256;
  *(uint4*)(s + (i0 >> 2) * LDSTR + (i0 & 3) * 8) = r[0];
  *(uint4*)(s + (i1 >> 2) * LDSTR + (i1 & 3) * 8) = r[1];
}

#if USE_ASYNC_LDS
__device__ __forceinline__ void issue_tile_async(const __bf16* __restrict__ G, long K,
                                                 long rowBase, long kbase, int t,
                                                 __bf16* sbuf) {
  const int i0 = t, i1 = t + 256;
  async_cp16(G + (rowBase + (long)(i0 >> 2)) * K + kbase + (long)(i0 & 3) * 8,
             sbuf + (i0 >> 2) * LDSTR + (i0 & 3) * 8);
  async_cp16(G + (rowBase + (long)(i1 >> 2)) * K + kbase + (long)(i1 & 3) * 8,
             sbuf + (i1 >> 2) * LDSTR + (i1 & 3) * 8);
}
#endif

// ---------------------------------------------------------------------------
// GEMM: C[M,N] = A[M,K] (bf16) * W[N,K]^T (bf16) + bias[N]
// EPI: 0 = fp32 store, 1 = bf16 store, 2 = bf16 store with ReLU
// Block tile 128x128x32, 8 waves, each wave 64x32 (4x2 WMMA tiles).
// Requires M%128==0, N%128==0, K%32==0 (true for all call sites).
// ---------------------------------------------------------------------------
template <int EPI>
__global__ __launch_bounds__(256)
void gemm_bf16_nt(const __bf16* __restrict__ A, const __bf16* __restrict__ W,
                  const float* __restrict__ bias, void* __restrict__ Cout,
                  int M, int N, int K) {
  __shared__ __bf16 sA[2][128 * LDSTR];
  __shared__ __bf16 sB[2][128 * LDSTR];
  const int t    = threadIdx.x;
  const int lane = t & 31;
  const int wid  = t >> 5;
  const int wm   = wid >> 2;      // 0..1 : 64-row slab
  const int wn   = wid & 3;       // 0..3 : 32-col slab
  const int hf   = lane >> 4;
  const int l16  = lane & 15;
  const long rowBase = (long)blockIdx.y * 128;
  const long colBase = (long)blockIdx.x * 128;

  v8f acc[4][2];
#pragma unroll
  for (int i = 0; i < 4; ++i)
#pragma unroll
    for (int j = 0; j < 2; ++j)
#pragma unroll
      for (int e = 0; e < 8; ++e) acc[i][j][e] = 0.f;

  const int KT = K >> 5;

#if USE_ASYNC_LDS
  // ---- async double-buffered pipeline: no VGPR data staging ----
  issue_tile_async(A, K, rowBase, 0, t, sA[0]);
  issue_tile_async(W, K, colBase, 0, t, sB[0]);
  for (int kt = 0; kt < KT; ++kt) {
    wait_async0();            // this wave's tile-kt async copies have landed
    __syncthreads();          // => every wave's copies landed; prev compute done
    if (kt + 1 < KT) {        // overlap tile kt+1 loads with tile kt compute
      issue_tile_async(A, K, rowBase, (long)(kt + 1) * 32, t, sA[(kt + 1) & 1]);
      issue_tile_async(W, K, colBase, (long)(kt + 1) * 32, t, sB[(kt + 1) & 1]);
    }
    const __bf16* pa = &sA[kt & 1][(wm * 64 + l16) * LDSTR];
    const __bf16* pb = &sB[kt & 1][(wn * 32 + l16) * LDSTR];
    v16bf bf0 = load_frag(pb, hf);
    v16bf bf1 = load_frag(pb + 16 * LDSTR, hf);
#pragma unroll
    for (int i = 0; i < 4; ++i) {
      v16bf af = load_frag(pa + i * 16 * LDSTR, hf);
      acc[i][0] = __builtin_amdgcn_wmma_f32_16x16x32_bf16(false, af, false, bf0,
                                                          (short)0, acc[i][0], false, false);
      acc[i][1] = __builtin_amdgcn_wmma_f32_16x16x32_bf16(false, af, false, bf1,
                                                          (short)0, acc[i][1], false, false);
    }
  }
#else
  // ---- fallback: register-staged double buffer ----
  uint4 ra[2], rb[2];
  stage_load(A, K, rowBase, 0, t, ra);
  stage_load(W, K, colBase, 0, t, rb);
  stage_store(sA[0], t, ra);
  stage_store(sB[0], t, rb);
  for (int kt = 0; kt < KT; ++kt) {
    __syncthreads();
    if (kt + 1 < KT) {
      stage_load(A, K, rowBase, (long)(kt + 1) * 32, t, ra);
      stage_load(W, K, colBase, (long)(kt + 1) * 32, t, rb);
    }
    const __bf16* pa = &sA[kt & 1][(wm * 64 + l16) * LDSTR];
    const __bf16* pb = &sB[kt & 1][(wn * 32 + l16) * LDSTR];
    v16bf bf0 = load_frag(pb, hf);
    v16bf bf1 = load_frag(pb + 16 * LDSTR, hf);
#pragma unroll
    for (int i = 0; i < 4; ++i) {
      v16bf af = load_frag(pa + i * 16 * LDSTR, hf);
      acc[i][0] = __builtin_amdgcn_wmma_f32_16x16x32_bf16(false, af, false, bf0,
                                                          (short)0, acc[i][0], false, false);
      acc[i][1] = __builtin_amdgcn_wmma_f32_16x16x32_bf16(false, af, false, bf1,
                                                          (short)0, acc[i][1], false, false);
    }
    if (kt + 1 < KT) {
      stage_store(sA[(kt + 1) & 1], t, ra);
      stage_store(sB[(kt + 1) & 1], t, rb);
    }
  }
#endif

  // Epilogue. C/D layout: lane = {hf, n=l16}, VGPR e -> row e + 8*hf.
#pragma unroll
  for (int j = 0; j < 2; ++j) {
    const long col = colBase + wn * 32 + j * 16 + l16;
    const float bv = bias[col];
#pragma unroll
    for (int i = 0; i < 4; ++i) {
#pragma unroll
      for (int e = 0; e < 8; ++e) {
        const long row = rowBase + wm * 64 + i * 16 + e + 8 * hf;
        float v = acc[i][j][e] + bv;
        if (EPI == 2) v = fmaxf(v, 0.f);
        if (EPI == 0) ((float*)Cout)[row * N + col] = v;
        else          ((__bf16*)Cout)[row * N + col] = (__bf16)v;
      }
    }
  }
}

// ---------------------------------------------------------------------------
// Embedding: x = sin(concat(src, predict) * (k+1)); write fp32 master + bf16.
// ---------------------------------------------------------------------------
__global__ void embed_kernel(const float* __restrict__ src, const float* __restrict__ pred,
                             float* __restrict__ X, __bf16* __restrict__ Xh) {
  const long idx = (long)blockIdx.x * 256 + threadIdx.x;
  if (idx >= (long)MROWS * DMODEL) return;
  const int  k   = (int)(idx & 511);
  const long row = idx >> 9;
  const float x  = (k < 511) ? src[row * 511 + k] : pred[row];
  const float y  = sinf(x * (float)(k + 1));
  X[idx]  = y;
  Xh[idx] = (__bf16)y;
}

__global__ void cvt_kernel(const float* __restrict__ w, __bf16* __restrict__ o, int n) {
  const int i = blockIdx.x * 256 + threadIdx.x;
  if (i < n) o[i] = (__bf16)w[i];
}

// ---------------------------------------------------------------------------
// Attention (nhead=1, S=64). One block per sequence n (512 blocks).
// QKV bf16 [32768][1536]; token (s,n) at row s*512+n; q/k/v at col 0/512/1024.
// Only ~2% of layer FLOPs -> fp32 VALU with LDS staging, fused softmax.
// ---------------------------------------------------------------------------
__global__ __launch_bounds__(256)
void attn_kernel(const __bf16* __restrict__ QKV, __bf16* __restrict__ O) {
  const int n  = blockIdx.x;
  const int t  = threadIdx.x;
  const int r  = t >> 2;       // score row 0..63 owned by 4 consecutive lanes
  const int qq = t & 3;

  __shared__ __bf16 smem[64 * 72 * 2];   // stage1: Q | K  ; stage2: V chunk [64][136]
  __shared__ __bf16 sAm[64 * 72];        // softmax(A) in bf16
  __bf16* sQ = smem;
  __bf16* sK = smem + 64 * 72;
  __bf16* sV = smem;

  float acc[16];
#pragma unroll
  for (int i = 0; i < 16; ++i) acc[i] = 0.f;

  // scores = q @ k^T over K=512 in chunks of 64
  for (int kc = 0; kc < 8; ++kc) {
    const long base = ((long)(r * 512 + n)) * 1536 + kc * 64 + qq * 16;
    const uint4* gq = (const uint4*)(QKV + base);
    const uint4* gk = (const uint4*)(QKV + base + 512);
    uint4* dq = (uint4*)(sQ + r * 72 + qq * 16);
    uint4* dk = (uint4*)(sK + r * 72 + qq * 16);
    dq[0] = gq[0]; dq[1] = gq[1];
    dk[0] = gk[0]; dk[1] = gk[1];
    __syncthreads();
#pragma unroll 4
    for (int c = 0; c < 16; ++c) {
      const int cc = qq * 16 + c;
      float sum = 0.f;
      for (int e = 0; e < 64; ++e)
        sum += (float)sQ[r * 72 + e] * (float)sK[cc * 72 + e];
      acc[c] += sum;
    }
    __syncthreads();
  }

  // scale + softmax across the 4 lanes sharing row r
  const float scale = 0.04419417382415922f;   // 1/sqrt(512)
  float mx = -1e30f;
#pragma unroll
  for (int c = 0; c < 16; ++c) { acc[c] *= scale; mx = fmaxf(mx, acc[c]); }
  mx = fmaxf(mx, __shfl_xor(mx, 1, 32));
  mx = fmaxf(mx, __shfl_xor(mx, 2, 32));
  float sum = 0.f;
#pragma unroll
  for (int c = 0; c < 16; ++c) { acc[c] = __expf(acc[c] - mx); sum += acc[c]; }
  sum += __shfl_xor(sum, 1, 32);
  sum += __shfl_xor(sum, 2, 32);
  const float inv = 1.f / sum;
#pragma unroll
  for (int c = 0; c < 16; ++c) sAm[r * 72 + qq * 16 + c] = (__bf16)(acc[c] * inv);
  __syncthreads();

  // O = A @ V, 128 columns of V per chunk staged in LDS
  for (int cchunk = 0; cchunk < 4; ++cchunk) {
    __syncthreads();
    const long vbase = ((long)(r * 512 + n)) * 1536 + 1024 + cchunk * 128 + qq * 32;
    const uint4* g = (const uint4*)(QKV + vbase);
    uint4* d = (uint4*)(sV + r * 136 + qq * 32);
    d[0] = g[0]; d[1] = g[1]; d[2] = g[2]; d[3] = g[3];
    __syncthreads();
    for (int j = 0; j < 32; ++j) {
      float s = 0.f;
      for (int s2 = 0; s2 < 64; ++s2)
        s += (float)sAm[r * 72 + s2] * (float)sV[s2 * 136 + qq * 32 + j];
      O[((long)(r * 512 + n)) * 512 + cchunk * 128 + qq * 32 + j] = (__bf16)s;
    }
  }
}

// ---------------------------------------------------------------------------
// Fused residual add + LayerNorm (one wave per token row, shuffle reductions).
// X (fp32 master) updated in place; Xh gets the bf16 copy for the next GEMM.
// ---------------------------------------------------------------------------
__global__ __launch_bounds__(256)
void add_ln_kernel(float* __restrict__ X, const float* __restrict__ Y,
                   const float* __restrict__ g, const float* __restrict__ b,
                   __bf16* __restrict__ Xh) {
  const int  lane = threadIdx.x & 31;
  const int  wid  = threadIdx.x >> 5;
  const long row  = (long)blockIdx.x * 8 + wid;
  const long base = row * DMODEL;
  float v[16];
  float s = 0.f;
#pragma unroll
  for (int i = 0; i < 16; ++i) {
    const int e = i * 32 + lane;
    v[i] = X[base + e] + Y[base + e];
    s += v[i];
  }
#pragma unroll
  for (int o = 16; o >= 1; o >>= 1) s += __shfl_xor(s, o, 32);
  const float m = s * (1.f / 512.f);
  float s2 = 0.f;
#pragma unroll
  for (int i = 0; i < 16; ++i) { const float d = v[i] - m; s2 += d * d; }
#pragma unroll
  for (int o = 16; o >= 1; o >>= 1) s2 += __shfl_xor(s2, o, 32);
  const float rstd = rsqrtf(s2 * (1.f / 512.f) + 1e-5f);
#pragma unroll
  for (int i = 0; i < 16; ++i) {
    const int e = i * 32 + lane;
    const float o = (v[i] - m) * rstd * g[e] + b[e];
    X[base + e]  = o;
    Xh[base + e] = (__bf16)o;
  }
}

// ---------------------------------------------------------------------------
// Per-batch 512x512 transpose; writes fp32 + bf16 copies.
// ---------------------------------------------------------------------------
__global__ __launch_bounds__(256)
void transpose_kernel(const float* __restrict__ X, float* __restrict__ XT,
                      __bf16* __restrict__ XTh) {
  __shared__ float tile[32][33];
  const int b  = blockIdx.z;
  const int x0 = blockIdx.x * 32, y0 = blockIdx.y * 32;
  const int tx = threadIdx.x & 31, ty = threadIdx.x >> 5;
  const float* src = X + (long)b * 512 * 512;
#pragma unroll
  for (int k = 0; k < 4; ++k)
    tile[ty + 8 * k][tx] = src[(long)(y0 + ty + 8 * k) * 512 + x0 + tx];
  __syncthreads();
  float*  dst  = XT  + (long)b * 512 * 512;
  __bf16* dsth = XTh + (long)b * 512 * 512;
#pragma unroll
  for (int k = 0; k < 4; ++k) {
    const float v = tile[tx][ty + 8 * k];
    const long  o = (long)(x0 + ty + 8 * k) * 512 + y0 + tx;
    dst[o] = v;
    dsth[o] = (__bf16)v;
  }
}

// mean over middle axis: out[b][j] = mean_i X[b][i][j]
__global__ __launch_bounds__(256)
void mean_kernel(const float* __restrict__ X, float* __restrict__ Mn) {
  const int b = blockIdx.x, t = threadIdx.x;
  for (int jj = 0; jj < 2; ++jj) {
    const int j = t + jj * 256;
    float s = 0.f;
    for (int i = 0; i < 512; ++i) s += X[((long)(b * 512 + i)) * 512 + j];
    Mn[b * 512 + j] = s * (1.f / 512.f);
  }
}

// head: out[b][c] = Mn[b] . lin_W[c] + lin_b[c]
__global__ void final_kernel(const float* __restrict__ Mn, const float* __restrict__ Wl,
                             const float* __restrict__ bl, float* __restrict__ out) {
  const int t = threadIdx.x;
  if (t >= 192) return;
  const int b = t / 3, c = t % 3;
  float s = 0.f;
  for (int j = 0; j < 512; ++j) s += Mn[b * 512 + j] * Wl[c * 512 + j];
  out[t] = s + bl[c];
}

// ---------------------------------------------------------------------------
// Host orchestration
// ---------------------------------------------------------------------------
static void run_layer(hipStream_t stream, float* X, __bf16* Xh,
                      const __bf16* Wqkv, const float* bqkv,
                      const __bf16* Wo,   const float* bo,
                      const __bf16* W1,   const float* b1,
                      const __bf16* W2,   const float* b2,
                      const float* g1, const float* be1,
                      const float* g2, const float* be2,
                      __bf16* QKVh, __bf16* Oh, __bf16* Hh, float* Y) {
  const dim3 blk(256);
  gemm_bf16_nt<1><<<dim3(12, 256), blk, 0, stream>>>(Xh, Wqkv, bqkv, QKVh, MROWS, 1536, 512);
  attn_kernel<<<512, 256, 0, stream>>>(QKVh, Oh);
  gemm_bf16_nt<0><<<dim3(4, 256), blk, 0, stream>>>(Oh, Wo, bo, Y, MROWS, 512, 512);
  add_ln_kernel<<<4096, 256, 0, stream>>>(X, Y, g1, be1, Xh);
  gemm_bf16_nt<2><<<dim3(16, 256), blk, 0, stream>>>(Xh, W1, b1, Hh, MROWS, 2048, 512);
  gemm_bf16_nt<0><<<dim3(4, 256), blk, 0, stream>>>(Hh, W2, b2, Y, MROWS, 512, 2048);
  add_ln_kernel<<<4096, 256, 0, stream>>>(X, Y, g2, be2, Xh);
}

extern "C" void kernel_launch(void* const* d_in, const int* in_sizes, int n_in,
                              void* d_out, int out_size, void* d_ws, size_t ws_size,
                              hipStream_t stream) {
  (void)in_sizes; (void)n_in; (void)out_size; (void)ws_size;

  const float* src  = (const float*)d_in[0];
  const float* pred = (const float*)d_in[1];
  // d_in[2] = masks (all zeros, unused by the reference math)
  const float* hW[4] = {(const float*)d_in[3], (const float*)d_in[5],
                        (const float*)d_in[7], (const float*)d_in[9]};
  const float* hb[4] = {(const float*)d_in[4], (const float*)d_in[6],
                        (const float*)d_in[8], (const float*)d_in[10]};
  const float* hln[4] = {(const float*)d_in[11], (const float*)d_in[12],
                         (const float*)d_in[13], (const float*)d_in[14]};
  const float* vW[4] = {(const float*)d_in[15], (const float*)d_in[17],
                        (const float*)d_in[19], (const float*)d_in[21]};
  const float* vb[4] = {(const float*)d_in[16], (const float*)d_in[18],
                        (const float*)d_in[20], (const float*)d_in[22]};
  const float* vln[4] = {(const float*)d_in[23], (const float*)d_in[24],
                         (const float*)d_in[25], (const float*)d_in[26]};
  const float* linW = (const float*)d_in[27];
  const float* linb = (const float*)d_in[28];

  // --- workspace carve (256-byte aligned regions) ---
  char* w = (char*)d_ws;
  size_t off = 0;
  auto carve = [&](size_t bytes) -> void* {
    void* p = w + off;
    off += (bytes + 255) & ~(size_t)255;
    return p;
  };
  const long MD = (long)MROWS * DMODEL;
  float*  X    = (float*)carve(MD * 4);
  float*  X2   = (float*)carve(MD * 4);
  __bf16* Xh   = (__bf16*)carve(MD * 2);
  float*  Y    = (float*)carve(MD * 4);
  __bf16* QKVh = (__bf16*)carve((long)MROWS * 1536 * 2);
  __bf16* Oh   = (__bf16*)carve(MD * 2);
  __bf16* Hh   = (__bf16*)carve((long)MROWS * 2048 * 2);
  __bf16* Wbuf = (__bf16*)carve((size_t)2 * 3145728 * 2);
  float*  Mn   = (float*)carve((size_t)64 * 512 * 4);

  const int wsz[4] = {1536 * 512, 512 * 512, 2048 * 512, 512 * 2048};
  __bf16* hWh[4]; __bf16* vWh[4];
  {
    size_t o2 = 0;
    for (int i = 0; i < 4; ++i) { hWh[i] = Wbuf + o2; o2 += wsz[i]; }
    for (int i = 0; i < 4; ++i) { vWh[i] = Wbuf + o2; o2 += wsz[i]; }
  }

  // 1) embedding
  embed_kernel<<<(unsigned)((MD + 255) / 256), 256, 0, stream>>>(src, pred, X, Xh);

  // 2) weights -> bf16
  for (int i = 0; i < 4; ++i) {
    cvt_kernel<<<(wsz[i] + 255) / 256, 256, 0, stream>>>(hW[i], hWh[i], wsz[i]);
    cvt_kernel<<<(wsz[i] + 255) / 256, 256, 0, stream>>>(vW[i], vWh[i], wsz[i]);
  }

  // 3) three "h" layers
  for (int l = 0; l < 3; ++l)
    run_layer(stream, X, Xh, hWh[0], hb[0], hWh[1], hb[1], hWh[2], hb[2], hWh[3], hb[3],
              hln[0], hln[1], hln[2], hln[3], QKVh, Oh, Hh, Y);

  // 4) transpose (0,2,1) per batch; X2 becomes the new fp32 master
  transpose_kernel<<<dim3(16, 16, 64), 256, 0, stream>>>(X, X2, Xh);

  // 5) three "v" layers
  for (int l = 0; l < 3; ++l)
    run_layer(stream, X2, Xh, vWh[0], vb[0], vWh[1], vb[1], vWh[2], vb[2], vWh[3], vb[3],
              vln[0], vln[1], vln[2], vln[3], QKVh, Oh, Hh, Y);

  // 6) mean over axis 1, then linear head
  mean_kernel<<<64, 256, 0, stream>>>(X2, Mn);
  final_kernel<<<1, 192, 0, stream>>>(Mn, linW, linb, (float*)d_out);
}
